// SparseBlockSeries_37288906063942
// MI455X (gfx1250) — matrix operand order repaired and checked by reading, not compiled, not run on hardware
//
#include <hip/hip_runtime.h>

typedef _Float16 v8h  __attribute__((ext_vector_type(8)));
typedef _Float16 v16h __attribute__((ext_vector_type(16)));
typedef float    v8f  __attribute__((ext_vector_type(8)));

#define KOFF 27
#define CCH  64
#define BN_EPS 1e-4f

// ---------------------------------------------------------------------------
// Prep: W [2,2,27,64,64] fp32 (k,c,d) -> f16 transposed (k,d,c) so each B
// fragment is one contiguous 32B load per lane in the WMMA B layout.
// ---------------------------------------------------------------------------
__global__ void prep_w_kernel(const float* __restrict__ w,
                              _Float16* __restrict__ wh, int total) {
  int t = blockIdx.x * blockDim.x + threadIdx.x;
  if (t >= total) return;
  int c   = t & 63;
  int d   = (t >> 6) & 63;
  int ijk = t >> 12;                       // 0..107 = (i*2+j)*27+k
  wh[(ijk << 12) + (d << 6) + c] = (_Float16)w[(ijk << 12) + (c << 6) + d];
}

// feats fp32 -> xh f16 with zero pad row N; also zero pad row of hh.
__global__ void prep_x_kernel(const float* __restrict__ feats,
                              _Float16* __restrict__ xh,
                              _Float16* __restrict__ hh, int N) {
  int t = blockIdx.x * blockDim.x + threadIdx.x;
  int total = (N + 1) * CCH;
  if (t >= total) return;
  if (t < N * CCH) {
    xh[t] = (_Float16)feats[t];
  } else {                                 // pad row (missing-neighbor target)
    xh[t] = (_Float16)0.f;
    hh[t] = (_Float16)0.f;
  }
}

// ---------------------------------------------------------------------------
// Submanifold conv as gather-GEMM on WMMA.
// One wave -> TWO 16-row M-tiles (32 rows) x 64 output channels; each B
// fragment is reused for both M-tiles (halves weight traffic from L2).
// 27k x 2ksteps x 4ntiles x 2mtiles = 432 WMMA per wave.
// xh: [N+1,64] f16 (row N = zeros), wh: [27,64(d),64(c)] f16, y: [N,64] fp32.
// ---------------------------------------------------------------------------
__global__ __launch_bounds__(256)
void subm_conv_wmma_kernel(const _Float16* __restrict__ xh,
                           const int* __restrict__ nbrs,
                           const _Float16* __restrict__ wh,
                           float* __restrict__ y, int N) {
  const int lane = threadIdx.x & 31;
  const int wave = threadIdx.x >> 5;
  const int tile0 = (blockIdx.x * 8 + wave) << 5;   // 32 rows per wave
  if (tile0 >= N) return;                           // wave-uniform: EXEC stays all-1s

  const int mrow = lane & 15;                       // A row / B column (N) index
  const int hi   = lane >> 4;                       // half-wave selector
  const int grow0 = tile0 + mrow;                   // M-tile 0 gather row
  const int grow1 = grow0 + 16;                     // M-tile 1 gather row
  const bool gv0 = grow0 < N;
  const bool gv1 = grow1 < N;
  const int* nrow0 = nbrs + (size_t)(gv0 ? grow0 : 0) * KOFF;
  const int* nrow1 = nbrs + (size_t)(gv1 ? grow1 : 0) * KOFF;

  v8f acc0[4] = {};                                 // rows tile0..tile0+15
  v8f acc1[4] = {};                                 // rows tile0+16..tile0+31

  for (int k = 0; k < KOFF; ++k) {
    const int r0 = gv0 ? nrow0[k] : N;              // N = padded zero row
    const int r1 = gv1 ? nrow1[k] : N;
    const _Float16* xr0 = xh + (size_t)r0 * CCH;
    const _Float16* xr1 = xh + (size_t)r1 * CCH;
    const _Float16* wk  = wh + (k << 12) + (mrow << 6);
#pragma unroll
    for (int kk = 0; kk < 2; ++kk) {                // K-steps of 32 over c=0..63
      // A fragments (16x32 f16): lane holds K = hi*16 .. +15 contiguous
      const int cbA = (kk << 5) + (hi << 4);
      v8h a0lo = *(const v8h*)(xr0 + cbA);
      v8h a0hi = *(const v8h*)(xr0 + cbA + 8);
      v8h a1lo = *(const v8h*)(xr1 + cbA);
      v8h a1hi = *(const v8h*)(xr1 + cbA + 8);
      v16h a0 = __builtin_shufflevector(a0lo, a0hi,
                  0,1,2,3,4,5,6,7,8,9,10,11,12,13,14,15);
      v16h a1 = __builtin_shufflevector(a1lo, a1hi,
                  0,1,2,3,4,5,6,7,8,9,10,11,12,13,14,15);
      // B fragment (32x16 f16): lane n=mrow, K = kk*32 + hi*16 .. +15 contiguous
      const _Float16* wb = wk + (kk << 5) + (hi << 4);
#pragma unroll
      for (int t = 0; t < 4; ++t) {                 // 4 column tiles of 16
        v16h b = *(const v16h*)(wb + (t << 10));    // +t*16*64 halfs
        acc0[t] = __builtin_amdgcn_wmma_f32_16x16x32_f16(
            false, a0, false, b, (short)0, acc0[t], false, false);
        acc1[t] = __builtin_amdgcn_wmma_f32_16x16x32_f16(
            false, a1, false, b, (short)0, acc1[t], false, false);
      }
    }
  }

  // C/D layout: VGPR j -> M = hi*8 + j, N = mrow
  float* yr = y + (size_t)tile0 * CCH;
#pragma unroll
  for (int t = 0; t < 4; ++t) {
#pragma unroll
    for (int j = 0; j < 8; ++j) {
      const int row0 = (hi << 3) + j;
      const int row1 = row0 + 16;
      if (tile0 + row0 < N)
        yr[row0 * CCH + (t << 4) + mrow] = acc0[t][j];
      if (tile0 + row1 < N)
        yr[row1 * CCH + (t << 4) + mrow] = acc1[t][j];
    }
  }
}

// ---------------------------------------------------------------------------
// BN stats: per-channel sum & sum-of-squares (population var), LDS-reduced,
// finished with 2x64 global f32 atomics per block. stats[0..63]=sum, [64..127]=sumsq.
// ---------------------------------------------------------------------------
__global__ __launch_bounds__(256)
void bn_stats_kernel(const float* __restrict__ y, float* __restrict__ stats, int N) {
  __shared__ float s_sum[256];
  __shared__ float s_sq[256];
  const int c   = threadIdx.x & 63;
  const int sub = threadIdx.x >> 6;                 // 4 rows in flight per block
  float s = 0.f, s2 = 0.f;
  for (int row = blockIdx.x * 4 + sub; row < N; row += gridDim.x * 4) {
    float v = y[(size_t)row * CCH + c];
    s += v; s2 += v * v;
  }
  s_sum[threadIdx.x] = s;
  s_sq[threadIdx.x]  = s2;
  __syncthreads();
  if (sub == 0) {
    float ts = s_sum[c] + s_sum[c + 64] + s_sum[c + 128] + s_sum[c + 192];
    float t2 = s_sq[c]  + s_sq[c + 64]  + s_sq[c + 128]  + s_sq[c + 192];
    atomicAdd(&stats[c], ts);
    atomicAdd(&stats[64 + c], t2);
  }
}

// BN + ReLU, emit f16 for next conv.
__global__ void bn_relu_f16_kernel(const float* __restrict__ y,
                                   const float* __restrict__ stats,
                                   const float* __restrict__ gamma,
                                   const float* __restrict__ beta,
                                   _Float16* __restrict__ out_h, int N) {
  int t = blockIdx.x * blockDim.x + threadIdx.x;
  if (t >= N * CCH) return;
  int c = t & 63;
  float inv_n = 1.f / (float)N;
  float mu  = stats[c] * inv_n;
  float var = stats[64 + c] * inv_n - mu * mu;
  float v = (y[t] - mu) * rsqrtf(var + BN_EPS) * gamma[c] + beta[c];
  out_h[t] = (_Float16)fmaxf(v, 0.f);
}

// BN + residual add + ReLU, emit both fp32 (next residual / final out) and f16.
__global__ void bn_add_relu_kernel(const float* __restrict__ z,
                                   const float* __restrict__ stats,
                                   const float* __restrict__ gamma,
                                   const float* __restrict__ beta,
                                   const float* __restrict__ residual,
                                   float* __restrict__ xout,
                                   _Float16* __restrict__ xout_h, int N) {
  int t = blockIdx.x * blockDim.x + threadIdx.x;
  if (t >= N * CCH) return;
  int c = t & 63;
  float inv_n = 1.f / (float)N;
  float mu  = stats[c] * inv_n;
  float var = stats[64 + c] * inv_n - mu * mu;
  float v = (z[t] - mu) * rsqrtf(var + BN_EPS) * gamma[c] + beta[c] + residual[t];
  v = fmaxf(v, 0.f);
  xout[t]   = v;
  xout_h[t] = (_Float16)v;
}

// ---------------------------------------------------------------------------
extern "C" void kernel_launch(void* const* d_in, const int* in_sizes, int n_in,
                              void* d_out, int out_size, void* d_ws, size_t ws_size,
                              hipStream_t stream) {
  const float* feats = (const float*)d_in[0];
  const int*   nbrs  = (const int*)d_in[1];
  const float* W     = (const float*)d_in[2];
  const float* gamma = (const float*)d_in[3];
  const float* beta  = (const float*)d_in[4];
  const int N = in_sizes[0] / CCH;

  // Workspace carve (256B aligned)
  char* ws = (char*)d_ws;
  size_t off = 0;
  auto carve = [&](size_t bytes) -> void* {
    void* p = ws + off;
    off += (bytes + 255) & ~(size_t)255;
    return p;
  };
  const int wtotal = 2 * 2 * KOFF * CCH * CCH;                    // 442368
  _Float16* wh    = (_Float16*)carve((size_t)wtotal * sizeof(_Float16));
  _Float16* xh    = (_Float16*)carve((size_t)(N + 1) * CCH * sizeof(_Float16));
  _Float16* hh    = (_Float16*)carve((size_t)(N + 1) * CCH * sizeof(_Float16));
  float*    ybuf  = (float*)carve((size_t)N * CCH * sizeof(float));
  float*    xbuf  = (float*)carve((size_t)N * CCH * sizeof(float));
  float*    stats = (float*)carve(128 * sizeof(float));

  prep_w_kernel<<<(wtotal + 255) / 256, 256, 0, stream>>>(W, wh, wtotal);
  prep_x_kernel<<<((N + 1) * CCH + 255) / 256, 256, 0, stream>>>(feats, xh, hh, N);

  const int convGrid = (N + 255) / 256;            // 8 waves x 32 rows per block
  const int elemGrid = (N * CCH + 255) / 256;

  for (int i = 0; i < 2; ++i) {
    const _Float16* wh0 = wh + (size_t)(i * 2 + 0) * KOFF * CCH * CCH;
    const _Float16* wh1 = wh + (size_t)(i * 2 + 1) * KOFF * CCH * CCH;
    const float* g0 = gamma + (size_t)(i * 2 + 0) * CCH;
    const float* b0 = beta  + (size_t)(i * 2 + 0) * CCH;
    const float* g1 = gamma + (size_t)(i * 2 + 1) * CCH;
    const float* b1 = beta  + (size_t)(i * 2 + 1) * CCH;

    // conv1 -> BN stats -> BN+ReLU (f16)
    subm_conv_wmma_kernel<<<convGrid, 256, 0, stream>>>(xh, nbrs, wh0, ybuf, N);
    hipMemsetAsync(stats, 0, 128 * sizeof(float), stream);
    bn_stats_kernel<<<512, 256, 0, stream>>>(ybuf, stats, N);
    bn_relu_f16_kernel<<<elemGrid, 256, 0, stream>>>(ybuf, stats, g0, b0, hh, N);

    // conv2 -> BN stats -> BN + residual + ReLU
    subm_conv_wmma_kernel<<<convGrid, 256, 0, stream>>>(hh, nbrs, wh1, ybuf, N);
    hipMemsetAsync(stats, 0, 128 * sizeof(float), stream);
    bn_stats_kernel<<<512, 256, 0, stream>>>(ybuf, stats, N);

    const float* resid = (i == 0) ? feats : xbuf;
    float* xo = (i == 1) ? (float*)d_out : xbuf;
    bn_add_relu_kernel<<<elemGrid, 256, 0, stream>>>(ybuf, stats, g1, b1, resid,
                                                     xo, xh, N);
  }
}
